// AttentionBlock_64561948393808
// MI455X (gfx1250) — matrix operand order repaired and checked by reading, not compiled, not run on hardware
//
#include <hip/hip_runtime.h>
#include <math.h>

#define TOKENS  4096
#define SEQ     2048
#define BATCH   2
#define DMODEL  1024
#define DFF     4096
#define NHEAD   16
#define HDIM    64

typedef __attribute__((ext_vector_type(16))) __bf16 v16bf;
typedef __attribute__((ext_vector_type(8)))  float  v8f;

union Frag {
    v16bf v;
    unsigned short u[16];
    uint4 q[2];
};

__device__ __forceinline__ unsigned short f2bf(float f) {
    unsigned int u = __float_as_uint(f);
    u += 0x7FFFu + ((u >> 16) & 1u);
    return (unsigned short)(u >> 16);
}

__device__ __forceinline__ v8f v8f_zero() {
    v8f z = {0.f, 0.f, 0.f, 0.f, 0.f, 0.f, 0.f, 0.f};
    return z;
}

// A-fragment: lane row = r, halves split K as {8*hi..8*hi+7, 16+8*hi..+7}
__device__ __forceinline__ void loadA(Frag& f, const unsigned short* row, int kk, int hi) {
    f.q[0] = *(const uint4*)(row + kk + 8 * hi);
    f.q[1] = *(const uint4*)(row + kk + 16 + 8 * hi);
}
// B-fragment: lane col = r, halves split K as {8*hi..8*hi+7, 16+8*hi..+7}
__device__ __forceinline__ void loadB(Frag& f, const unsigned short* row, int kk, int hi) {
    f.q[0] = *(const uint4*)(row + kk + 8 * hi);
    f.q[1] = *(const uint4*)(row + kk + 16 + 8 * hi);
}

// ---------------------------------------------------------------- weight cast
__global__ void cast_f32_bf16_kernel(const float* __restrict__ in,
                                     unsigned short* __restrict__ out, int n) {
    int i = blockIdx.x * blockDim.x + threadIdx.x;
    int stride = gridDim.x * blockDim.x;
    for (; i < n; i += stride) out[i] = f2bf(in[i]);
}

// ---------------------------------------------------------------- layernorm -> bf16
__global__ void ln_bf16_kernel(const float* __restrict__ x,
                               const float* __restrict__ g,
                               const float* __restrict__ b,
                               unsigned short* __restrict__ out) {
    __shared__ float s1[256];
    __shared__ float s2[256];
    int t = blockIdx.x;
    const float* xr = x + (size_t)t * DMODEL;
    float xs[4];
    float sum = 0.f, sq = 0.f;
#pragma unroll
    for (int i = 0; i < 4; i++) {
        float v = xr[threadIdx.x + i * 256];
        xs[i] = v; sum += v; sq += v * v;
    }
    s1[threadIdx.x] = sum; s2[threadIdx.x] = sq;
    __syncthreads();
    for (int o = 128; o > 0; o >>= 1) {
        if ((int)threadIdx.x < o) {
            s1[threadIdx.x] += s1[threadIdx.x + o];
            s2[threadIdx.x] += s2[threadIdx.x + o];
        }
        __syncthreads();
    }
    float mu  = s1[0] * (1.0f / DMODEL);
    float var = s2[0] * (1.0f / DMODEL) - mu * mu;
    float rs  = rsqrtf(var + 1e-5f);
    unsigned short* orow = out + (size_t)t * DMODEL;
#pragma unroll
    for (int i = 0; i < 4; i++) {
        int d = threadIdx.x + i * 256;
        orow[d] = f2bf((xs[i] - mu) * rs * g[d] + b[d]);
    }
}

// ---------------------------------------------------------------- RoPE + repack
__global__ void rope_pack_kernel(const float* __restrict__ qf,
                                 const float* __restrict__ kf,
                                 const float* __restrict__ vf,
                                 unsigned short* __restrict__ qbf,
                                 unsigned short* __restrict__ kbf,
                                 unsigned short* __restrict__ vT) {
    int idx = blockIdx.x * blockDim.x + threadIdx.x;
    if (idx >= TOKENS * DMODEL) return;
    int token = idx >> 10;
    int c = idx & 1023;
    int h = c >> 6;
    int d = c & 63;
    int b = token / SEQ;
    int s = token - b * SEQ;
    int i = d & 31;
    float inv = powf(10000.0f, -(float)(2 * i) / 64.0f);
    float ang = (float)s * inv;
    float cs = cosf(ang), sn = sinf(ang);
    float qv = qf[idx], kv = kf[idx];
    float qo = (d < 32) ? -qf[idx + 32] : qf[idx - 32];
    float ko = (d < 32) ? -kf[idx + 32] : kf[idx - 32];
    size_t hbase = (size_t)(b * NHEAD + h);
    qbf[(hbase * SEQ + s) * HDIM + d] = f2bf(qv * cs + qo * sn);
    kbf[(hbase * SEQ + s) * HDIM + d] = f2bf(kv * cs + ko * sn);
    vT[(hbase * HDIM + d) * SEQ + s]  = f2bf(vf[idx]);
}

// ---------------------------------------------------------------- bf16 WMMA GEMM
// C[m,n] = sum_k A[m,k]*Bw[n,k] + bias[n]  (+ epilogue)
// 32x64 output per wave; full k-step double buffering: the next k-step's
// 2 A-fragments and all 4 B-fragments are issued before the current 8 WMMAs.
// EPI: 0 = f32 out, 1 = f32 out + residual, 2 = bf16 gelu out, 3 = f32 + res + mask
template <int EPI>
__global__ __launch_bounds__(128, 1)
void gemm_bf16_kernel(const unsigned short* __restrict__ A,
                      const unsigned short* __restrict__ Bw,
                      const float* __restrict__ bias,
                      const float* __restrict__ res,
                      const unsigned char* __restrict__ mask,
                      void* __restrict__ outv,
                      int M, int N, int K) {
    int lane = threadIdx.x & 31;
    int wave = threadIdx.x >> 5;
    int r  = lane & 15;
    int hi = lane >> 4;
    int m0 = blockIdx.x * 32;
    int n0 = blockIdx.y * 256 + wave * 64;

    const unsigned short* arow0 = A + (size_t)(m0 + r) * K;
    const unsigned short* arow1 = A + (size_t)(m0 + 16 + r) * K;
    const unsigned short* brow  = Bw + (size_t)(n0 + r) * K;   // + nt*16*K per N-tile

    v8f acc[2][4];
#pragma unroll
    for (int mt = 0; mt < 2; mt++)
#pragma unroll
        for (int nt = 0; nt < 4; nt++) acc[mt][nt] = v8f_zero();

    Frag aF[2][2];      // [m-tile][phase buffer]
    Frag bF[4][2];      // [n-tile][phase buffer]

    loadA(aF[0][0], arow0, 0, hi);
    loadA(aF[1][0], arow1, 0, hi);
#pragma unroll
    for (int nt = 0; nt < 4; nt++) loadB(bF[nt][0], brow + (size_t)nt * 16 * K, 0, hi);

    int iters2 = K / 64;   // K is a multiple of 64
    for (int it2 = 0; it2 < iters2; it2++) {
#pragma unroll
        for (int ph = 0; ph < 2; ph++) {
            int kk = (it2 * 2 + ph) * 32;
            int nb = ph ^ 1;
            bool have_next = (kk + 32 < K);
            if (have_next) {
                loadA(aF[0][nb], arow0, kk + 32, hi);
                loadA(aF[1][nb], arow1, kk + 32, hi);
#pragma unroll
                for (int nt = 0; nt < 4; nt++)
                    loadB(bF[nt][nb], brow + (size_t)nt * 16 * K, kk + 32, hi);
            }
#pragma unroll
            for (int nt = 0; nt < 4; nt++) {
                acc[0][nt] = __builtin_amdgcn_wmma_f32_16x16x32_bf16(
                    false, aF[0][ph].v, false, bF[nt][ph].v, (short)0, acc[0][nt], false, false);
                acc[1][nt] = __builtin_amdgcn_wmma_f32_16x16x32_bf16(
                    false, aF[1][ph].v, false, bF[nt][ph].v, (short)0, acc[1][nt], false, false);
            }
        }
    }

#pragma unroll
    for (int nt = 0; nt < 4; nt++) {
        int n = n0 + nt * 16 + r;
        float bn = bias[n];
#pragma unroll
        for (int mt = 0; mt < 2; mt++) {
#pragma unroll
            for (int v = 0; v < 8; v++) {
                int m = m0 + mt * 16 + v + 8 * hi;
                size_t oi = (size_t)m * N + n;
                float val = acc[mt][nt][v] + bn;
                if (EPI == 0) {
                    ((float*)outv)[oi] = val;
                } else if (EPI == 1) {
                    ((float*)outv)[oi] = val + res[oi];
                } else if (EPI == 2) {
                    float gl = 0.5f * val * (1.0f + erff(val * 0.70710678118f));
                    ((unsigned short*)outv)[oi] = f2bf(gl);
                } else {
                    float o = val + res[oi];
                    if (!mask[m]) o = 0.0f;
                    ((float*)outv)[oi] = o;
                }
            }
        }
    }
}

// ---------------------------------------------------------------- flash attention
// one wave per (batch, head, 16-query tile); 32 keys per iteration.
// K-fragments have short live ranges (consumed by score WMMAs immediately);
// V-fragments are hoisted so their latency hides under the softmax VALU work.
__global__ __launch_bounds__(128, 1)
void flash_attn_kernel(const unsigned short* __restrict__ qbf,
                       const unsigned short* __restrict__ kbf,
                       const unsigned short* __restrict__ vT,
                       const unsigned char* __restrict__ mask,
                       unsigned short* __restrict__ ao) {
    __shared__ __align__(16) unsigned short pbuf[4][16 * 32];
    int lane  = threadIdx.x & 31;
    int wslot = threadIdx.x >> 5;
    int wid = blockIdx.x * 4 + wslot;
    int qt = wid & 127;           // S/16 = 128 query tiles
    int hh = (wid >> 7) & 15;
    int b  = wid >> 11;
    int r  = lane & 15;
    int hi = lane >> 4;

    size_t headoff = (size_t)(b * NHEAD + hh) * SEQ * HDIM;
    const unsigned short* qrow = qbf + headoff + (size_t)(qt * 16 + r) * HDIM;
    const unsigned short* kh   = kbf + headoff;
    const unsigned short* vh   = vT + (size_t)(b * NHEAD + hh) * HDIM * SEQ;

    Frag aq[2];
#pragma unroll
    for (int c = 0; c < 2; c++) {
        aq[c].q[0] = *(const uint4*)(qrow + c * 32 + 8 * hi);
        aq[c].q[1] = *(const uint4*)(qrow + c * 32 + 16 + 8 * hi);
    }

    float mstat[8], lstat[8];
#pragma unroll
    for (int v = 0; v < 8; v++) { mstat[v] = -__builtin_inff(); lstat[v] = 0.0f; }
    v8f o[4];
#pragma unroll
    for (int nt = 0; nt < 4; nt++) o[nt] = v8f_zero();

    const float scale = 0.125f;   // 1/sqrt(64)

    for (int j = 0; j < SEQ / 32; j++) {
        // prefetch next 32-key block of K and V
        if (j + 1 < SEQ / 32) {
            __builtin_prefetch(kh + (size_t)((j + 1) * 32 + lane) * HDIM, 0, 3);
            __builtin_prefetch(vh + (size_t)lane * SEQ + (j + 1) * 32, 0, 3);
            __builtin_prefetch(vh + (size_t)(lane + 32) * SEQ + (j + 1) * 32, 0, 3);
        }

        // V fragments: independent of softmax, issued early, consumed last
        Frag bv[4];
#pragma unroll
        for (int nt = 0; nt < 4; nt++) {
            const unsigned short* vrow = vh + (size_t)(nt * 16 + r) * SEQ + j * 32 + 8 * hi;
            bv[nt].q[0] = *(const uint4*)(vrow);
            bv[nt].q[1] = *(const uint4*)(vrow + 16);
        }

        // scores: K fragments consumed immediately by the WMMAs
        v8f sc[2];
#pragma unroll
        for (int t = 0; t < 2; t++) {
            int key = j * 32 + t * 16 + r;
            const unsigned short* krow = kh + (size_t)key * HDIM;
            Frag bk0, bk1;
            bk0.q[0] = *(const uint4*)(krow + 8 * hi);
            bk0.q[1] = *(const uint4*)(krow + 16 + 8 * hi);
            bk1.q[0] = *(const uint4*)(krow + 32 + 8 * hi);
            bk1.q[1] = *(const uint4*)(krow + 32 + 16 + 8 * hi);
            v8f a = v8f_zero();
            a = __builtin_amdgcn_wmma_f32_16x16x32_bf16(
                false, aq[0].v, false, bk0.v, (short)0, a, false, false);
            a = __builtin_amdgcn_wmma_f32_16x16x32_bf16(
                false, aq[1].v, false, bk1.v, (short)0, a, false, false);
            sc[t] = a;
        }

        bool mk0 = mask[b * SEQ + j * 32 + r] != 0;
        bool mk1 = mask[b * SEQ + j * 32 + 16 + r] != 0;

#pragma unroll
        for (int v = 0; v < 8; v++) {
            float s0 = mk0 ? sc[0][v] * scale : -__builtin_inff();
            float s1 = mk1 ? sc[1][v] * scale : -__builtin_inff();
            float mx = fmaxf(s0, s1);
            mx = fmaxf(mx, __shfl_xor(mx, 1, 16));
            mx = fmaxf(mx, __shfl_xor(mx, 2, 16));
            mx = fmaxf(mx, __shfl_xor(mx, 4, 16));
            mx = fmaxf(mx, __shfl_xor(mx, 8, 16));
            float mnew = fmaxf(mstat[v], mx);
            float alpha, p0, p1;
            if (mnew == -__builtin_inff()) {
                alpha = 1.0f; p0 = 0.0f; p1 = 0.0f;
            } else {
                alpha = expf(mstat[v] - mnew);
                p0 = expf(s0 - mnew);
                p1 = expf(s1 - mnew);
            }
            float rs = p0 + p1;
            rs += __shfl_xor(rs, 1, 16);
            rs += __shfl_xor(rs, 2, 16);
            rs += __shfl_xor(rs, 4, 16);
            rs += __shfl_xor(rs, 8, 16);
            lstat[v] = lstat[v] * alpha + rs;
            mstat[v] = mnew;
#pragma unroll
            for (int nt = 0; nt < 4; nt++) o[nt][v] *= alpha;
            pbuf[wslot][(v + 8 * hi) * 32 + r]      = f2bf(p0);
            pbuf[wslot][(v + 8 * hi) * 32 + 16 + r] = f2bf(p1);
        }

        // redistribute P from C-layout to A-layout through LDS
        Frag pf;
        pf.q[0] = *(const uint4*)&pbuf[wslot][r * 32 + 8 * hi];
        pf.q[1] = *(const uint4*)&pbuf[wslot][r * 32 + 16 + 8 * hi];

#pragma unroll
        for (int nt = 0; nt < 4; nt++) {
            o[nt] = __builtin_amdgcn_wmma_f32_16x16x32_bf16(
                false, pf.v, false, bv[nt].v, (short)0, o[nt], false, false);
        }
    }

#pragma unroll
    for (int nt = 0; nt < 4; nt++) {
#pragma unroll
        for (int v = 0; v < 8; v++) {
            int m = v + 8 * hi;
            int srow = qt * 16 + m;
            float denom = lstat[v];
            float val = (denom > 0.0f) ? o[nt][v] / denom : 0.0f;
            ao[(size_t)(b * SEQ + srow) * DMODEL + hh * HDIM + nt * 16 + r] = f2bf(val);
        }
    }
}

// ---------------------------------------------------------------- launcher
extern "C" void kernel_launch(void* const* d_in, const int* in_sizes, int n_in,
                              void* d_out, int out_size, void* d_ws, size_t ws_size,
                              hipStream_t stream) {
    (void)in_sizes; (void)n_in; (void)out_size; (void)ws_size;
    const float* x     = (const float*)d_in[0];
    const unsigned char* mask = (const unsigned char*)d_in[1];
    const float* ln1_g = (const float*)d_in[2];
    const float* ln1_b = (const float*)d_in[3];
    const float* Wq = (const float*)d_in[4];  const float* bq = (const float*)d_in[5];
    const float* Wk = (const float*)d_in[6];  const float* bk = (const float*)d_in[7];
    const float* Wv = (const float*)d_in[8];  const float* bv = (const float*)d_in[9];
    const float* Wo = (const float*)d_in[10]; const float* bo = (const float*)d_in[11];
    const float* ln2_g = (const float*)d_in[12];
    const float* ln2_b = (const float*)d_in[13];
    const float* W1 = (const float*)d_in[14]; const float* b1 = (const float*)d_in[15];
    const float* W2 = (const float*)d_in[16]; const float* b2 = (const float*)d_in[17];
    float* out = (float*)d_out;
    char* ws = (char*)d_ws;
    const size_t MB = 1024u * 1024u;

    unsigned short* wq_bf = (unsigned short*)(ws + 0 * MB);
    unsigned short* wk_bf = (unsigned short*)(ws + 2 * MB);
    unsigned short* wv_bf = (unsigned short*)(ws + 4 * MB);
    unsigned short* wo_bf = (unsigned short*)(ws + 6 * MB);
    unsigned short* w1_bf = (unsigned short*)(ws + 8 * MB);
    unsigned short* w2_bf = (unsigned short*)(ws + 16 * MB);
    unsigned short* y_bf  = (unsigned short*)(ws + 24 * MB);   // later reused as h_bf
    float* qf = (float*)(ws + 32 * MB);                        // later reused as y1f
    float* kf = (float*)(ws + 48 * MB);                        // later reused by h1_bf
    float* vf = (float*)(ws + 64 * MB);                        // later reused by h1_bf
    unsigned short* qbf  = (unsigned short*)(ws + 80 * MB);
    unsigned short* kbf  = (unsigned short*)(ws + 88 * MB);
    unsigned short* vTbf = (unsigned short*)(ws + 96 * MB);
    unsigned short* aobf = (unsigned short*)(ws + 104 * MB);
    float* y1f = qf;                                           // overlay (qf dead)
    unsigned short* h_bf  = y_bf;                              // overlay (y_bf dead)
    unsigned short* h1_bf = (unsigned short*)(ws + 48 * MB);   // overlay (kf,vf dead)

    // 1) weights -> bf16
    cast_f32_bf16_kernel<<<2048, 256, 0, stream>>>(Wq, wq_bf, DMODEL * DMODEL);
    cast_f32_bf16_kernel<<<2048, 256, 0, stream>>>(Wk, wk_bf, DMODEL * DMODEL);
    cast_f32_bf16_kernel<<<2048, 256, 0, stream>>>(Wv, wv_bf, DMODEL * DMODEL);
    cast_f32_bf16_kernel<<<2048, 256, 0, stream>>>(Wo, wo_bf, DMODEL * DMODEL);
    cast_f32_bf16_kernel<<<4096, 256, 0, stream>>>(W1, w1_bf, DFF * DMODEL);
    cast_f32_bf16_kernel<<<4096, 256, 0, stream>>>(W2, w2_bf, DMODEL * DFF);

    // 2) LN1 -> bf16
    ln_bf16_kernel<<<TOKENS, 256, 0, stream>>>(x, ln1_g, ln1_b, y_bf);

    // 3) QKV projections (bf16 WMMA, fp32 out)
    dim3 gqkv(TOKENS / 32, DMODEL / 256);
    gemm_bf16_kernel<0><<<gqkv, 128, 0, stream>>>(y_bf, wq_bf, bq, nullptr, nullptr, qf, TOKENS, DMODEL, DMODEL);
    gemm_bf16_kernel<0><<<gqkv, 128, 0, stream>>>(y_bf, wk_bf, bk, nullptr, nullptr, kf, TOKENS, DMODEL, DMODEL);
    gemm_bf16_kernel<0><<<gqkv, 128, 0, stream>>>(y_bf, wv_bf, bv, nullptr, nullptr, vf, TOKENS, DMODEL, DMODEL);

    // 4) RoPE + repack (V transposed per head)
    rope_pack_kernel<<<(TOKENS * DMODEL) / 256, 256, 0, stream>>>(qf, kf, vf, qbf, kbf, vTbf);

    // 5) flash attention: B*H*(S/16) = 4096 waves, 4 waves/block
    flash_attn_kernel<<<1024, 128, 0, stream>>>(qbf, kbf, vTbf, mask, aobf);

    // 6) output projection + residual
    gemm_bf16_kernel<1><<<gqkv, 128, 0, stream>>>(aobf, wo_bf, bo, x, nullptr, y1f, TOKENS, DMODEL, DMODEL);

    // 7) LN2 -> bf16
    ln_bf16_kernel<<<TOKENS, 256, 0, stream>>>(y1f, ln2_g, ln2_b, h_bf);

    // 8) FFN up + exact GELU -> bf16
    dim3 gffn1(TOKENS / 32, DFF / 256);
    gemm_bf16_kernel<2><<<gffn1, 128, 0, stream>>>(h_bf, w1_bf, b1, nullptr, nullptr, h1_bf, TOKENS, DFF, DMODEL);

    // 9) FFN down + residual + mask -> fp32 output
    dim3 gffn2(TOKENS / 32, DMODEL / 256);
    gemm_bf16_kernel<3><<<gffn2, 128, 0, stream>>>(h1_bf, w2_bf, b2, y1f, mask, out, TOKENS, DMODEL, DFF);
}
